// Net_63471026700733
// MI455X (gfx1250) — matrix operand order repaired and checked by reading, not compiled, not run on hardware
//
#include <hip/hip_runtime.h>
#include <hip/hip_bf16.h>
#include <math.h>

typedef _Float16 f16;
typedef __attribute__((ext_vector_type(16))) _Float16 v16h;
typedef __attribute__((ext_vector_type(8)))  _Float16 v8h;
typedef __attribute__((ext_vector_type(8)))  float    v8f;

// ---------------- problem dims ----------------
#define NB   16        // batch
#define NT   1600      // raw time
#define ND   80        // raw feat
#define NL   6         // layers
#define NH   4         // heads
#define DM   256       // model dim
#define DKH  64        // head dim
#define NE   4         // experts
#define HID  1024
#define EMB  512
#define NV   4233
#define NVP  4240      // vocab dim padded to multiple of 8 (16B f16 chunks)
#define TP   399       // subsampled time
#define TPP  416       // padded time stride (multiple of 8) for probs/kT/pT
#define FS   19        // subsampled freq
#define T1   799       // after conv1 (time)
#define F1   39        // after conv1 (freq)
#define MTOK (NB*TP)   // 6384 tokens

// ---------------- GEMM epilogue modes ----------------
enum {
  M_F32_BIAS_SCALE = 0, // outF = (acc+bias)*scale            (subsample linear)
  M_H_BIAS,             // outH = acc+bias                    (conv pw1)
  M_H_BIAS_SWISH,       // outH = swish(acc+bias)             (MoE hidden)
  M_MOE_ACC,            // if sel==e: x += 0.5*gval*(acc+b2)  (MoE out)
  M_QUQV,               // q+bq -> (+u)->qu, (+v)->qv  (b,h,t,d)
  M_KT,                 // k+bk -> kT (b,h,d,t) ld TPP
  M_VV,                 // v+bv -> vv (b,h,t,d)
  M_PT,                 // pos proj -> pT (h*64+d, t) ld TPP
  M_SCORES,             // dual-K: (qu@kT + qv@pT)*0.125 -> f32
  M_CTX,                // probs@v -> att_o (b,t, h*64+d) f16
  M_ACCUM,              // x += acc+bias                      (o-proj)
  M_PW2,                // x += mask ? acc+bias : 0           (conv pw2)
  M_OUT,                // d_out = acc+bias                   (vocab proj)
  M_CONV2               // implicit-GEMM conv2, relu, scatter to (b,t,c*19+f)
};

struct GemmP {
  const f16* A;  const f16* Bw;
  const f16* A2; const f16* B2w;
  long lda, ldb;
  long strideAz, strideBz, strideB2z;
  int  b2mod;
  int  M, N, K, K2;
  const float* bias; const float* bias2; const float* bias3;
  float scale;
  float* outF; f16* outH; f16* outH2;
  long ldo;
  const int* sel; const float* gsel; int expert;
  const int* lens;
};

// =====================================================================
// WMMA GEMM: 64x64 block tile, 4 waves, each wave 32x32 (2x2 wmma accs),
// K staged 64 per barrier (2 WMMA K-substeps).
//  - Global->LDS staging uses 128-bit loads (v8h) when lda/ldb are
//    multiples of 8 halves (all call sites except the conv2 gather).
//  - A tile row-major in LDS; B tile stored TRANSPOSED (n-major, k
//    contiguous) so both A and B fragments are assembled with contiguous
//    128-bit ds loads. Row stride 72 halves = 144B keeps 16B alignment.
// Fragment layouts per CDNA5 ISA 7.12.2 (16-bit A 16x32, B 32x16, f32 C).
// =====================================================================
template<int MODE>
__global__ __launch_bounds__(128) void gemm_wmma(GemmP p)
{
  constexpr bool DUAL   = (MODE == M_SCORES);
  constexpr bool GATHER = (MODE == M_CONV2);
  __shared__ __align__(16) f16 As[64][72];   // [m][k]
  __shared__ __align__(16) f16 Bs[64][72];   // [n][k]  (transposed stage)

  const int tid  = threadIdx.x;
  const int wave = tid >> 5, lane = tid & 31;
  const int wm = wave >> 1, wn = wave & 1;
  const int z  = blockIdx.z;
  const int m0 = blockIdx.y * 64;
  const int n0 = blockIdx.x * 64;
  const int mlane = lane & 15, hi = lane >> 4;

  const f16* Ab  = p.A  + (size_t)z * p.strideAz;
  const f16* Bb  = p.Bw + (size_t)z * p.strideBz;
  const f16* A2b = DUAL ? (p.A2  + (size_t)z * p.strideAz) : (const f16*)0;
  const f16* B2b = DUAL ? (p.B2w + (size_t)(p.b2mod ? (z % p.b2mod) : z) * p.strideB2z)
                        : (const f16*)0;

  const bool vecA = !GATHER && ((p.lda & 7) == 0);
  const bool vecB = ((p.ldb & 7) == 0);

  v8f acc[2][2] = {};

  for (int phase = 0; phase < (DUAL ? 2 : 1); ++phase) {
    const f16* Ap = (phase == 0) ? Ab : A2b;
    const f16* Bp = (phase == 0) ? Bb : B2b;
    const int  Kp = (phase == 0) ? p.K : p.K2;

    for (int k0 = 0; k0 < Kp; k0 += 64) {
      // ---- stage A tile (64 rows x 64 k) into LDS ----
      if (vecA) {
        for (int idx = tid; idx < 512; idx += 128) {
          int r = idx >> 3, c8 = (idx & 7) * 8;
          int gm = m0 + r, gk = k0 + c8;
          v8h v = {};
          if (gm < p.M) {
            if (gk + 8 <= Kp) {
              v = *(const v8h*)(Ap + (size_t)gm * p.lda + gk);
            } else {
              #pragma unroll
              for (int e = 0; e < 8; ++e)
                if (gk + e < Kp) v[e] = Ap[(size_t)gm * p.lda + gk + e];
            }
          }
          *(v8h*)&As[r][c8] = v;
        }
      } else {
        for (int idx = tid; idx < 64 * 64; idx += 128) {
          int r = idx >> 6, c = idx & 63;
          int gm = m0 + r, gk = k0 + c;
          f16 v = (f16)0.f;
          if (gm < p.M && gk < Kp) {
            if (GATHER) { // implicit im2col for conv2: row=(b,to,fo), k=(kt,kf,ci)
              int bq = gm / (TP * FS);
              int rr = gm - bq * (TP * FS);
              int to = rr / FS, fo = rr - to * FS;
              int kt = gk / (3 * DM);
              int rem = gk - kt * (3 * DM);
              int kf = rem >> 8;
              int ci = rem & 255;
              v = Ap[(((size_t)bq * T1 + (2 * to + kt)) * F1 + (2 * fo + kf)) * DM + ci];
            } else {
              v = Ap[(size_t)gm * p.lda + gk];
            }
          }
          As[r][c] = v;
        }
      }
      // ---- stage B tile (64 k x 64 n) transposed into LDS ----
      if (vecB) {
        for (int idx = tid; idx < 512; idx += 128) {
          int r = idx >> 3, c8 = (idx & 7) * 8;   // r=k, c8=n chunk
          int gk = k0 + r, gn = n0 + c8;
          v8h v = {};
          if (gk < Kp) {
            if (gn + 8 <= p.N) {
              v = *(const v8h*)(Bp + (size_t)gk * p.ldb + gn);
            } else {
              #pragma unroll
              for (int e = 0; e < 8; ++e)
                if (gn + e < p.N) v[e] = Bp[(size_t)gk * p.ldb + gn + e];
            }
          }
          #pragma unroll
          for (int e = 0; e < 8; ++e) Bs[c8 + e][r] = v[e];
        }
      } else {
        for (int idx = tid; idx < 64 * 64; idx += 128) {
          int r = idx >> 6, c = idx & 63;
          int gk = k0 + r, gn = n0 + c;
          Bs[c][r] = (gk < Kp && gn < p.N) ? Bp[(size_t)gk * p.ldb + gn] : (f16)0.f;
        }
      }
      if (!GATHER && (k0 + 64) < Kp) { // prefetch hint for next K stage
        __builtin_prefetch(Bp + (size_t)(k0 + 64) * p.ldb + n0, 0, 1);
      }
      __syncthreads();

      // ---- fragments + WMMA (two K=32 substeps) ----
      #pragma unroll
      for (int ks = 0; ks < 2; ++ks) {
        const int k8 = ks * 4;                   // v8h offset within row
        v16h af[2], bf[2];
        #pragma unroll
        for (int i = 0; i < 2; ++i) {
          const v8h* arow = (const v8h*)&As[wm * 32 + i * 16 + mlane][0];
          v8h lo = arow[k8 + hi];                // halves ks*32 + hi*8 .. +7
          v8h hh = arow[k8 + 2 + hi];            // halves ks*32 + 16 + hi*8 .. +7
          af[i] = __builtin_shufflevector(lo, hh, 0,1,2,3,4,5,6,7,8,9,10,11,12,13,14,15);
        }
        #pragma unroll
        for (int j = 0; j < 2; ++j) {
          const v8h* brow = (const v8h*)&Bs[wn * 32 + j * 16 + mlane][0];
          v8h lo = brow[k8 + hi];                // halves ks*32 + hi*8 .. +7
          v8h hh = brow[k8 + 2 + hi];            // halves ks*32 + 16 + hi*8 .. +7
          bf[j] = __builtin_shufflevector(lo, hh, 0,1,2,3,4,5,6,7,8,9,10,11,12,13,14,15);
        }
        #pragma unroll
        for (int i = 0; i < 2; ++i)
          #pragma unroll
          for (int j = 0; j < 2; ++j)
            acc[i][j] = __builtin_amdgcn_wmma_f32_16x16x32_f16(
                false, af[i], false, bf[j], (short)0, acc[i][j], false, false);
      }
      __syncthreads();
    }
  }

  // ---- epilogue ----
  #pragma unroll
  for (int i = 0; i < 2; ++i)
  for (int j = 0; j < 2; ++j) {
    int nG = n0 + wn * 32 + j * 16 + mlane;
    #pragma unroll
    for (int vv = 0; vv < 8; ++vv) {
      int mG = m0 + wm * 32 + i * 16 + hi * 8 + vv;
      if (mG >= p.M || nG >= p.N) continue;
      float a = acc[i][j][vv];
      if (MODE == M_F32_BIAS_SCALE) {
        p.outF[(size_t)mG * p.ldo + nG] = (a + p.bias[nG]) * p.scale;
      } else if (MODE == M_H_BIAS) {
        p.outH[(size_t)mG * p.ldo + nG] = (f16)(a + p.bias[nG]);
      } else if (MODE == M_H_BIAS_SWISH) {
        float h = a + p.bias[nG];
        p.outH[(size_t)mG * p.ldo + nG] = (f16)(h / (1.f + expf(-h)));
      } else if (MODE == M_MOE_ACC) {
        if (p.sel[mG] == p.expert) {
          float y = a + p.bias[nG];
          p.outF[(size_t)mG * p.ldo + nG] += 0.5f * p.gsel[mG] * y;
        }
      } else if (MODE == M_QUQV) {
        float q = a + p.bias[nG];
        int bq = mG / TP, t = mG - bq * TP;
        int h = nG >> 6, d = nG & 63;
        size_t off = (((size_t)(bq * NH + h)) * TP + t) * DKH + d;
        p.outH [off] = (f16)(q + p.bias2[nG]);
        p.outH2[off] = (f16)(q + p.bias3[nG]);
      } else if (MODE == M_KT) {
        int bq = mG / TP, t = mG - bq * TP;
        int h = nG >> 6, d = nG & 63;
        p.outH[(((size_t)(bq * NH + h)) * DKH + d) * TPP + t] = (f16)(a + p.bias[nG]);
      } else if (MODE == M_VV) {
        int bq = mG / TP, t = mG - bq * TP;
        int h = nG >> 6, d = nG & 63;
        p.outH[(((size_t)(bq * NH + h)) * TP + t) * DKH + d] = (f16)(a + p.bias[nG]);
      } else if (MODE == M_PT) {
        p.outH[(size_t)nG * TPP + mG] = (f16)a;
      } else if (MODE == M_SCORES) {
        p.outF[(size_t)z * (TP * TP) + (size_t)mG * TP + nG] = a * p.scale;
      } else if (MODE == M_CTX) {
        int bq = z >> 2, h = z & 3;
        p.outH[((size_t)(bq * TP + mG)) * DM + h * DKH + nG] = (f16)a;
      } else if (MODE == M_ACCUM) {
        p.outF[(size_t)mG * p.ldo + nG] += a + p.bias[nG];
      } else if (MODE == M_PW2) {
        int bq = mG / TP, t = mG - bq * TP;
        float y = (4 * t < p.lens[bq]) ? (a + p.bias[nG]) : 0.f;
        p.outF[(size_t)mG * p.ldo + nG] += y;
      } else if (MODE == M_OUT) {
        p.outF[(size_t)mG * p.ldo + nG] = a + p.bias[nG];
      } else if (MODE == M_CONV2) {
        int bq = mG / (TP * FS);
        int rr = mG - bq * (TP * FS);
        int to = rr / FS, fo = rr - to * FS;
        float y = a + p.bias[nG];
        p.outH[((size_t)(bq * TP + to)) * 4864 + (size_t)nG * FS + fo] = (f16)fmaxf(y, 0.f);
      }
    }
  }
}

// =====================================================================
// small helper kernels
// =====================================================================
__global__ void cvt_f32_f16(const float* __restrict__ s, f16* __restrict__ d, long n) {
  long i  = (long)blockIdx.x * blockDim.x + threadIdx.x;
  long st = (long)gridDim.x * blockDim.x;
  for (; i < n; i += st) d[i] = (f16)s[i];
}

// f32->f16 repack with destination row padding (for odd N weights)
__global__ void cvtpad_kernel(const float* __restrict__ s, f16* __restrict__ d,
                              int rows, int cols, int ldd) {
  long i  = (long)blockIdx.x * blockDim.x + threadIdx.x;
  long n  = (long)rows * cols;
  long st = (long)gridDim.x * blockDim.x;
  for (; i < n; i += st) {
    long r = i / cols, c = i - r * cols;
    d[r * (long)ldd + c] = (f16)s[i];
  }
}

// conv2 weight permute: (co,ci,kt,kf) -> [(kt*3+kf)*256+ci][co]
__global__ void permc2_kernel(const float* __restrict__ w, f16* __restrict__ o) {
  int i = blockIdx.x * 256 + threadIdx.x;
  if (i >= 256 * 2304) return;
  int co = i / 2304, r = i % 2304, ci = r / 9, kk = r % 9;
  o[((size_t)kk * 256 + ci) * 256 + co] = (f16)w[i];
}

// conv1: 1->256ch, 3x3 stride2, relu. out layout (b, t(799), f(39), c(256)) f16
__global__ void conv1_kernel(const float* __restrict__ xs, const float* __restrict__ w,
                             const float* __restrict__ bias, f16* __restrict__ x1) {
  int c = threadIdx.x;
  int idx = blockIdx.x;                  // (b,t,f)
  int b = idx / (T1 * F1);
  int r = idx % (T1 * F1);
  int t = r / F1, f = r % F1;
  float s = bias[c];
  #pragma unroll
  for (int kt = 0; kt < 3; ++kt)
    #pragma unroll
    for (int kf = 0; kf < 3; ++kf) {
      float xv = xs[((size_t)b * NT + (2 * t + kt)) * ND + (2 * f + kf)];
      s += xv * w[c * 9 + kt * 3 + kf];
    }
  x1[(size_t)idx * DM + c] = (f16)fmaxf(s, 0.f);
}

// sinusoidal rel-pos table (TP x DM) f16
__global__ void pe_kernel(f16* __restrict__ pe) {
  int t = blockIdx.x, i = threadIdx.x;   // i: 0..127
  float dv = expf((2.f * i) * (-logf(10000.f) / (float)DM));
  float a = (float)t * dv;
  pe[(size_t)t * DM + 2 * i]     = (f16)sinf(a);
  pe[(size_t)t * DM + 2 * i + 1] = (f16)cosf(a);
}

// layernorm, one wave per row of 256.  OM: 0=f16 out, 1=f32 out, 2=f16 out masked
template<int OM>
__global__ __launch_bounds__(128) void ln_kernel(const float* __restrict__ x,
                                                 const float* __restrict__ g,
                                                 const float* __restrict__ b,
                                                 f16* outH, float* outF,
                                                 const int* lens, int rows) {
  int row = blockIdx.x * 4 + (threadIdx.x >> 5);
  if (row >= rows) return;
  int lane = threadIdx.x & 31;
  const float* xr = x + (size_t)row * DM;
  float vals[8], s = 0.f;
  #pragma unroll
  for (int j = 0; j < 8; ++j) { vals[j] = xr[lane + j * 32]; s += vals[j]; }
  #pragma unroll
  for (int o = 16; o > 0; o >>= 1) s += __shfl_xor(s, o);
  float mean = s * (1.f / DM);
  float v = 0.f;
  #pragma unroll
  for (int j = 0; j < 8; ++j) { float d = vals[j] - mean; v += d * d; }
  #pragma unroll
  for (int o = 16; o > 0; o >>= 1) v += __shfl_xor(v, o);
  float inv = rsqrtf(v * (1.f / DM) + 1e-12f);
  bool zero = false;
  if (OM == 2) { int bq = row / TP, t = row - bq * TP; zero = (4 * t >= lens[bq]); }
  #pragma unroll
  for (int j = 0; j < 8; ++j) {
    int col = lane + j * 32;
    float y = g[col] * (vals[j] - mean) * inv + b[col];
    if (OM == 2 && zero) y = 0.f;
    if (OM == 1) outF[(size_t)row * DM + col] = y;
    else         outH[(size_t)row * DM + col] = (f16)y;
  }
}

// MoE gate: softmax(concat(xn,embed) @ gw) -> argmax idx + max prob
__global__ __launch_bounds__(128) void gate_kernel(const f16* __restrict__ xn,
                                                   const float* __restrict__ embed,
                                                   const float* __restrict__ gw,
                                                   int* __restrict__ sel,
                                                   float* __restrict__ gval, int rows) {
  int row = blockIdx.x * 4 + (threadIdx.x >> 5);
  if (row >= rows) return;
  int lane = threadIdx.x & 31;
  float a0 = 0, a1 = 0, a2 = 0, a3 = 0;
  const f16* xr = xn + (size_t)row * DM;
  for (int k = lane; k < DM; k += 32) {
    float xv = (float)xr[k];
    const float* gr = gw + (size_t)k * NE;
    a0 += xv * gr[0]; a1 += xv * gr[1]; a2 += xv * gr[2]; a3 += xv * gr[3];
  }
  const float* er = embed + (size_t)row * EMB;
  for (int k = lane; k < EMB; k += 32) {
    float ev = er[k];
    const float* gr = gw + (size_t)(DM + k) * NE;
    a0 += ev * gr[0]; a1 += ev * gr[1]; a2 += ev * gr[2]; a3 += ev * gr[3];
  }
  #pragma unroll
  for (int o = 16; o > 0; o >>= 1) {
    a0 += __shfl_xor(a0, o); a1 += __shfl_xor(a1, o);
    a2 += __shfl_xor(a2, o); a3 += __shfl_xor(a3, o);
  }
  if (lane == 0) {
    float m = fmaxf(fmaxf(a0, a1), fmaxf(a2, a3));
    float e0 = expf(a0 - m), e1 = expf(a1 - m), e2 = expf(a2 - m), e3 = expf(a3 - m);
    float s = e0 + e1 + e2 + e3;
    float p0 = e0 / s, p1 = e1 / s, p2 = e2 / s, p3 = e3 / s;
    int bi = 0; float bp = p0;
    if (p1 > bp) { bp = p1; bi = 1; }
    if (p2 > bp) { bp = p2; bi = 2; }
    if (p3 > bp) { bp = p3; bi = 3; }
    sel[row] = bi; gval[row] = bp;
  }
}

// masked softmax over s (399) for scores rows (b,h,t); probs f16 (ld TPP)
__global__ __launch_bounds__(128) void attn_softmax_kernel(const float* __restrict__ sc,
                                                           f16* __restrict__ pb,
                                                           const int* __restrict__ lens) {
  int row = blockIdx.x * 4 + (threadIdx.x >> 5);
  if (row >= 64 * TP) return;
  int lane = threadIdx.x & 31;
  int z = row / TP, bq = z >> 2;
  int len = lens[bq];
  const float* sr = sc + (size_t)row * TP;
  float v[13]; bool ok[13];
  float mx = -3.0e38f;
  #pragma unroll
  for (int j = 0; j < 13; ++j) {
    int s_ = lane + j * 32;
    bool in = s_ < TP;
    ok[j] = in && (4 * s_ < len);
    float val = in ? (ok[j] ? sr[s_] : -1.0e9f) : -3.0e38f;
    v[j] = val; mx = fmaxf(mx, val);
  }
  #pragma unroll
  for (int o = 16; o > 0; o >>= 1) mx = fmaxf(mx, __shfl_xor(mx, o));
  float sum = 0.f;
  #pragma unroll
  for (int j = 0; j < 13; ++j) {
    int s_ = lane + j * 32;
    float e = (s_ < TP) ? expf(v[j] - mx) : 0.f;
    v[j] = e; sum += e;
  }
  #pragma unroll
  for (int o = 16; o > 0; o >>= 1) sum += __shfl_xor(sum, o);
  float inv = 1.f / sum;
  #pragma unroll
  for (int j = 0; j < 13; ++j) {
    int s_ = lane + j * 32;
    if (s_ < TP) pb[(size_t)row * TPP + s_] = (f16)(ok[j] ? v[j] * inv : 0.f);
  }
}

// GLU: a*sigmoid(g) from pw1 output (m x 512) -> (m x 256) f16
__global__ void glu_kernel(const f16* __restrict__ pw1, f16* __restrict__ glu, int rows) {
  int i = blockIdx.x * 256 + threadIdx.x;
  if (i >= rows * DM) return;
  int m = i >> 8, c = i & 255;
  float a = (float)pw1[(size_t)m * 512 + c];
  float g = (float)pw1[(size_t)m * 512 + 256 + c];
  glu[i] = (f16)(a / (1.f + expf(-g)));
}

// depthwise conv k=15 pad=7 along time + bias + bn + swish
__global__ void dwconv_kernel(const f16* __restrict__ glu, const float* __restrict__ wdw,
                              const float* __restrict__ bdw, const float* __restrict__ bng,
                              const float* __restrict__ bnb, f16* __restrict__ out) {
  int c = threadIdx.x;
  int idx = blockIdx.x;                  // (b,t)
  int b = idx / TP, t = idx % TP;
  float s = bdw[c];
  #pragma unroll
  for (int j = 0; j < 15; ++j) {
    int tt = t + j - 7;
    if (tt >= 0 && tt < TP)
      s += (float)glu[((size_t)(b * TP + tt)) * DM + c] * wdw[c * 15 + j];
  }
  s = bng[c] * s + bnb[c];
  out[(size_t)idx * DM + c] = (f16)(s / (1.f + expf(-s)));
}

// out_lens = min(TP, ceil(len/4)), stored as float at tail of d_out
__global__ void lens_kernel(const int* __restrict__ lens, float* __restrict__ out) {
  int b = threadIdx.x;
  if (b < NB) {
    int v = (lens[b] + 3) >> 2;
    if (v > TP) v = TP;
    out[b] = (float)v;
  }
}

// =====================================================================
// host driver
// =====================================================================
extern "C" void kernel_launch(void* const* d_in, const int* in_sizes, int n_in,
                              void* d_out, int out_size, void* d_ws, size_t ws_size,
                              hipStream_t stream) {
  (void)in_sizes; (void)n_in; (void)out_size;
  // ------- inputs (setup_inputs dict order: xs, xs_lens, embed, params...) -------
  const float* xs     = (const float*)d_in[0];
  const int*   xl     = (const int*)  d_in[1];
  const float* embed  = (const float*)d_in[2];
  const float* c1w    = (const float*)d_in[3];
  const float* c1b    = (const float*)d_in[4];
  const float* c2w    = (const float*)d_in[5];
  const float* c2b    = (const float*)d_in[6];
  const float* slw    = (const float*)d_in[7];
  const float* slb    = (const float*)d_in[8];
  const float* ln_ffm_g = (const float*)d_in[9];
  const float* ln_ffm_b = (const float*)d_in[10];
  const float* ln_mha_g = (const float*)d_in[11];
  const float* ln_mha_b = (const float*)d_in[12];
  const float* ln_cv_g  = (const float*)d_in[13];
  const float* ln_cv_b  = (const float*)d_in[14];
  const float* ln_ff_g  = (const float*)d_in[15];
  const float* ln_ff_b  = (const float*)d_in[16];
  const float* ln_fin_g = (const float*)d_in[17];
  const float* ln_fin_b = (const float*)d_in[18];
  const float* mgate[2] = { (const float*)d_in[19], (const float*)d_in[24] };
  const float* mw1f [2] = { (const float*)d_in[20], (const float*)d_in[25] };
  const float* mb1  [2] = { (const float*)d_in[21], (const float*)d_in[26] };
  const float* mw2f [2] = { (const float*)d_in[22], (const float*)d_in[27] };
  const float* mb2  [2] = { (const float*)d_in[23], (const float*)d_in[28] };
  const float* wqf = (const float*)d_in[29]; const float* bq = (const float*)d_in[30];
  const float* wkf = (const float*)d_in[31]; const float* bk = (const float*)d_in[32];
  const float* wvf = (const float*)d_in[33]; const float* bv = (const float*)d_in[34];
  const float* wof = (const float*)d_in[35]; const float* bo = (const float*)d_in[36];
  const float* wposf = (const float*)d_in[37];
  const float* att_u = (const float*)d_in[38];
  const float* att_v = (const float*)d_in[39];
  const float* pw1f = (const float*)d_in[40]; const float* pw1b = (const float*)d_in[41];
  const float* dwwf = (const float*)d_in[42]; const float* dwbf = (const float*)d_in[43];
  const float* bngf = (const float*)d_in[44]; const float* bnbf = (const float*)d_in[45];
  const float* pw2f = (const float*)d_in[46]; const float* pw2b = (const float*)d_in[47];
  const float* olng = (const float*)d_in[48]; const float* olnb = (const float*)d_in[49];
  const float* owf  = (const float*)d_in[50]; const float* owb  = (const float*)d_in[51];

  // ------- workspace carving (~540 MB; assumes ws_size is sufficient) -------
  (void)ws_size;
  size_t off = 0;
  auto A16 = [&](size_t n) { f16* p = (f16*)((char*)d_ws + off);
                             off = (off + n * sizeof(f16) + 255) & ~(size_t)255; return p; };
  auto A32 = [&](size_t n) { float* p = (float*)((char*)d_ws + off);
                             off = (off + n * sizeof(float) + 255) & ~(size_t)255; return p; };
  auto AI  = [&](size_t n) { int* p = (int*)((char*)d_ws + off);
                             off = (off + n * sizeof(int) + 255) & ~(size_t)255; return p; };

  f16* wsubl = A16((size_t)4864 * DM);
  f16* wq = A16((size_t)NL * DM * DM);  f16* wk = A16((size_t)NL * DM * DM);
  f16* wv = A16((size_t)NL * DM * DM);  f16* wo = A16((size_t)NL * DM * DM);
  f16* wpos = A16((size_t)NL * DM * DM);
  f16* w1h[2] = { A16((size_t)NL * NE * DM * HID), A16((size_t)NL * NE * DM * HID) };
  f16* w2h[2] = { A16((size_t)NL * NE * HID * DM), A16((size_t)NL * NE * HID * DM) };
  f16* wpw1 = A16((size_t)NL * DM * 2 * DM);
  f16* wpw2 = A16((size_t)NL * DM * DM);
  f16* wout = A16((size_t)DM * NVP);          // padded ld
  f16* wc2  = A16((size_t)2304 * DM);
  f16* x1   = A16((size_t)NB * T1 * F1 * DM);
  f16* x2   = A16((size_t)MTOK * 4864);
  float* x  = A32((size_t)MTOK * DM);
  f16* xn   = A16((size_t)MTOK * DM);
  f16* pe16 = A16((size_t)TP * DM);
  f16* pT   = A16((size_t)DM * TPP);          // padded ld
  f16* qu   = A16((size_t)NB * NH * TP * DKH);
  f16* qv   = A16((size_t)NB * NH * TP * DKH);
  f16* kT   = A16((size_t)NB * NH * DKH * TPP);   // padded ld
  f16* vvB  = A16((size_t)NB * NH * TP * DKH);
  float* sco = A32((size_t)NB * NH * TP * TP);
  f16* prb  = A16((size_t)NB * NH * TP * TPP);    // padded ld
  f16* atto = A16((size_t)MTOK * DM);
  f16* hb   = A16((size_t)NE * MTOK * HID);
  f16* pw1o = A16((size_t)MTOK * 2 * DM);
  f16* glub = A16((size_t)MTOK * DM);
  f16* dwo  = A16((size_t)MTOK * DM);
  int* sel  = AI(MTOK);
  float* gvalb = A32(MTOK);

  auto cvt = [&](const float* s, f16* d, size_t n) {
    int blocks = (int)((n + 1023) / 1024); if (blocks > 4096) blocks = 4096;
    if (blocks < 1) blocks = 1;
    cvt_f32_f16<<<blocks, 256, 0, stream>>>(s, d, (long)n);
  };

  // ------- weight repack f32 -> f16 (KxN row-major, same contiguous layouts) -------
  cvt(slw, wsubl, (size_t)4864 * DM);
  cvt(wqf, wq, (size_t)NL * DM * DM);   cvt(wkf, wk, (size_t)NL * DM * DM);
  cvt(wvf, wv, (size_t)NL * DM * DM);   cvt(wof, wo, (size_t)NL * DM * DM);
  cvt(wposf, wpos, (size_t)NL * DM * DM);
  cvt(mw1f[0], w1h[0], (size_t)NL * NE * DM * HID);
  cvt(mw1f[1], w1h[1], (size_t)NL * NE * DM * HID);
  cvt(mw2f[0], w2h[0], (size_t)NL * NE * HID * DM);
  cvt(mw2f[1], w2h[1], (size_t)NL * NE * HID * DM);
  cvt(pw1f, wpw1, (size_t)NL * DM * 2 * DM);
  cvt(pw2f, wpw2, (size_t)NL * DM * DM);
  cvtpad_kernel<<<4096, 256, 0, stream>>>(owf, wout, DM, NV, NVP);
  permc2_kernel<<<(256 * 2304 + 255) / 256, 256, 0, stream>>>(c2w, wc2);

  // ------- subsampling -------
  conv1_kernel<<<NB * T1 * F1, 256, 0, stream>>>(xs, c1w, c1b, x1);

  { // conv2 as implicit-GEMM WMMA: M=(b,to,fo)=121296, N=256, K=2304
    GemmP p{}; p.A = x1; p.Bw = wc2; p.ldb = DM;
    p.M = NB * TP * FS; p.N = DM; p.K = 2304;
    p.bias = c2b; p.outH = x2;
    gemm_wmma<M_CONV2><<<dim3(DM / 64, (p.M + 63) / 64, 1), 128, 0, stream>>>(p);
  }
  { // x = (x2 @ sub_l + b) * sqrt(DM)
    GemmP p{}; p.A = x2; p.lda = 4864; p.Bw = wsubl; p.ldb = DM;
    p.M = MTOK; p.N = DM; p.K = 4864;
    p.bias = slb; p.scale = 16.f; p.outF = x; p.ldo = DM;
    gemm_wmma<M_F32_BIAS_SCALE><<<dim3(DM / 64, (MTOK + 63) / 64, 1), 128, 0, stream>>>(p);
  }
  pe_kernel<<<TP, 128, 0, stream>>>(pe16);

  const int lnBlocks = (MTOK + 3) / 4;

  // ------- conformer layers -------
  for (int l = 0; l < NL; ++l) {
    // ---- macaron MoE (m) and final MoE (f) share code ----
    for (int mod = 0; mod < 2; ++mod) {
      const float* g  = (mod == 0) ? ln_ffm_g : ln_ff_g;
      const float* bb = (mod == 0) ? ln_ffm_b : ln_ff_b;
      ln_kernel<0><<<lnBlocks, 128, 0, stream>>>(x, g + l * DM, bb + l * DM,
                                                 xn, nullptr, nullptr, MTOK);
      gate_kernel<<<lnBlocks, 128, 0, stream>>>(xn, embed, mgate[mod] + (size_t)l * 768 * NE,
                                                sel, gvalb, MTOK);
      for (int e = 0; e < NE; ++e) {
        GemmP p{}; p.A = xn; p.lda = DM;
        p.Bw = w1h[mod] + ((size_t)l * NE + e) * DM * HID; p.ldb = HID;
        p.M = MTOK; p.N = HID; p.K = DM;
        p.bias = mb1[mod] + ((size_t)l * NE + e) * HID;
        p.outH = hb + (size_t)e * MTOK * HID; p.ldo = HID;
        gemm_wmma<M_H_BIAS_SWISH><<<dim3(HID / 64, (MTOK + 63) / 64, 1), 128, 0, stream>>>(p);
      }
      for (int e = 0; e < NE; ++e) {
        GemmP p{}; p.A = hb + (size_t)e * MTOK * HID; p.lda = HID;
        p.Bw = w2h[mod] + ((size_t)l * NE + e) * HID * DM; p.ldb = DM;
        p.M = MTOK; p.N = DM; p.K = HID;
        p.bias = mb2[mod] + ((size_t)l * NE + e) * DM;
        p.outF = x; p.ldo = DM; p.sel = sel; p.gsel = gvalb; p.expert = e;
        gemm_wmma<M_MOE_ACC><<<dim3(DM / 64, (MTOK + 63) / 64, 1), 128, 0, stream>>>(p);
      }

      if (mod == 1) break; // after second MoE: fall through to ln_fin below

      // ---- attention ----
      ln_kernel<0><<<lnBlocks, 128, 0, stream>>>(x, ln_mha_g + l * DM, ln_mha_b + l * DM,
                                                 xn, nullptr, nullptr, MTOK);
      { GemmP p{}; p.A = xn; p.lda = DM; p.Bw = wq + (size_t)l * DM * DM; p.ldb = DM;
        p.M = MTOK; p.N = DM; p.K = DM;
        p.bias = bq + l * DM; p.bias2 = att_u + l * DM; p.bias3 = att_v + l * DM;
        p.outH = qu; p.outH2 = qv;
        gemm_wmma<M_QUQV><<<dim3(DM / 64, (MTOK + 63) / 64, 1), 128, 0, stream>>>(p); }
      { GemmP p{}; p.A = xn; p.lda = DM; p.Bw = wk + (size_t)l * DM * DM; p.ldb = DM;
        p.M = MTOK; p.N = DM; p.K = DM; p.bias = bk + l * DM; p.outH = kT;
        gemm_wmma<M_KT><<<dim3(DM / 64, (MTOK + 63) / 64, 1), 128, 0, stream>>>(p); }
      { GemmP p{}; p.A = xn; p.lda = DM; p.Bw = wv + (size_t)l * DM * DM; p.ldb = DM;
        p.M = MTOK; p.N = DM; p.K = DM; p.bias = bv + l * DM; p.outH = vvB;
        gemm_wmma<M_VV><<<dim3(DM / 64, (MTOK + 63) / 64, 1), 128, 0, stream>>>(p); }
      { GemmP p{}; p.A = pe16; p.lda = DM; p.Bw = wpos + (size_t)l * DM * DM; p.ldb = DM;
        p.M = TP; p.N = DM; p.K = DM; p.outH = pT;
        gemm_wmma<M_PT><<<dim3(DM / 64, (TP + 63) / 64, 1), 128, 0, stream>>>(p); }
      { // scores = (qu@kT + qv@pT)/8 ; z = b*H+h
        GemmP p{}; p.A = qu; p.A2 = qv; p.lda = DKH; p.strideAz = (long)TP * DKH;
        p.Bw = kT; p.ldb = TPP; p.strideBz = (long)DKH * TPP;
        p.B2w = pT; p.strideB2z = (long)DKH * TPP; p.b2mod = NH;
        p.M = TP; p.N = TP; p.K = DKH; p.K2 = DKH;
        p.scale = 0.125f; p.outF = sco;
        gemm_wmma<M_SCORES><<<dim3((TP + 63) / 64, (TP + 63) / 64, NB * NH), 128, 0, stream>>>(p); }
      attn_softmax_kernel<<<(64 * TP + 3) / 4, 128, 0, stream>>>(sco, prb, xl);
      { // context = probs @ v -> att_o
        GemmP p{}; p.A = prb; p.lda = TPP; p.strideAz = (long)TP * TPP;
        p.Bw = vvB; p.ldb = DKH; p.strideBz = (long)TP * DKH;
        p.M = TP; p.N = DKH; p.K = TP; p.outH = atto;
        gemm_wmma<M_CTX><<<dim3(1, (TP + 63) / 64, NB * NH), 128, 0, stream>>>(p); }
      { GemmP p{}; p.A = atto; p.lda = DM; p.Bw = wo + (size_t)l * DM * DM; p.ldb = DM;
        p.M = MTOK; p.N = DM; p.K = DM; p.bias = bo + l * DM;
        p.outF = x; p.ldo = DM;
        gemm_wmma<M_ACCUM><<<dim3(DM / 64, (MTOK + 63) / 64, 1), 128, 0, stream>>>(p); }

      // ---- conv module ----
      ln_kernel<2><<<lnBlocks, 128, 0, stream>>>(x, ln_cv_g + l * DM, ln_cv_b + l * DM,
                                                 xn, nullptr, xl, MTOK);
      { GemmP p{}; p.A = xn; p.lda = DM; p.Bw = wpw1 + (size_t)l * DM * 2 * DM; p.ldb = 2 * DM;
        p.M = MTOK; p.N = 2 * DM; p.K = DM; p.bias = pw1b + (size_t)l * 2 * DM;
        p.outH = pw1o; p.ldo = 2 * DM;
        gemm_wmma<M_H_BIAS><<<dim3(2 * DM / 64, (MTOK + 63) / 64, 1), 128, 0, stream>>>(p); }
      glu_kernel<<<(MTOK * DM + 255) / 256, 256, 0, stream>>>(pw1o, glub, MTOK);
      dwconv_kernel<<<MTOK, 256, 0, stream>>>(glub, dwwf + (size_t)l * DM * 15,
                                              dwbf + l * DM, bngf + l * DM, bnbf + l * DM, dwo);
      { GemmP p{}; p.A = dwo; p.lda = DM; p.Bw = wpw2 + (size_t)l * DM * DM; p.ldb = DM;
        p.M = MTOK; p.N = DM; p.K = DM; p.bias = pw2b + l * DM;
        p.outF = x; p.ldo = DM; p.lens = xl;
        gemm_wmma<M_PW2><<<dim3(DM / 64, (MTOK + 63) / 64, 1), 128, 0, stream>>>(p); }
    }
    // ---- per-layer final layernorm, in place on residual stream ----
    ln_kernel<1><<<lnBlocks, 128, 0, stream>>>(x, ln_fin_g + l * DM, ln_fin_b + l * DM,
                                               nullptr, x, nullptr, MTOK);
  }

  // ------- output head -------
  ln_kernel<0><<<lnBlocks, 128, 0, stream>>>(x, olng, olnb, xn, nullptr, nullptr, MTOK);
  { GemmP p{}; p.A = xn; p.lda = DM; p.Bw = wout; p.ldb = NVP;
    p.M = MTOK; p.N = NV; p.K = DM; p.bias = owb;
    p.outF = (float*)d_out; p.ldo = NV;
    gemm_wmma<M_OUT><<<dim3((NV + 63) / 64, (MTOK + 63) / 64, 1), 128, 0, stream>>>(p); }
  lens_kernel<<<1, 32, 0, stream>>>(xl, (float*)d_out + (size_t)MTOK * NV);
}
